// QLinear_32315333935635
// MI455X (gfx1250) — matrix operand order repaired and checked by reading, not compiled, not run on hardware
//
#include <hip/hip_runtime.h>
#include <stdint.h>

// ---------------- problem constants (from reference) ----------------
#define T_DIM   1024
#define IN_DIM  1024
#define OUT_DIM 1024
#define SUBD    128
#define NS      8          // IN / SUB
#define ZB      7          // activation bit slices
#define KS      7          // weight cell slices
#define QMAXI   127
#define LDA     136        // padded LDS row stride (halves) -> bank-conflict free

typedef __attribute__((ext_vector_type(16))) _Float16 v16h;
typedef __attribute__((ext_vector_type(8)))  _Float16 v8h;
typedef __attribute__((ext_vector_type(8)))  float    v8f;

// ---------------- helpers ----------------
__device__ __forceinline__ unsigned fenc(float f) {  // order-preserving float->uint
  unsigned u = __float_as_uint(f);
  return (u & 0x80000000u) ? ~u : (u | 0x80000000u);
}
__device__ __forceinline__ float fdec(unsigned u) {
  u = (u & 0x80000000u) ? (u & 0x7fffffffu) : ~u;
  return __uint_as_float(u);
}
__device__ __forceinline__ unsigned pcg(unsigned v) {
  v = v * 747796405u + 2891336453u;
  unsigned w = ((v >> ((v >> 28u) + 4u)) ^ v) * 277803737u;
  return (w >> 22u) ^ w;
}

// CDNA5 async global->LDS copy (bypasses VGPRs, tracked by ASYNCcnt).
// vdst = 32-bit LDS byte address (low 32 bits of generic shared pointer),
// vaddr = 64-bit global address, saddr = off.
__device__ __forceinline__ void async_g2l_b128(unsigned lds_addr, const void* gptr) {
  asm volatile("global_load_async_to_lds_b128 %0, %1, off"
               :: "v"(lds_addr), "v"((unsigned long long)(uintptr_t)gptr)
               : "memory");
}
__device__ __forceinline__ void wait_async0() {
#if __has_builtin(__builtin_amdgcn_s_wait_asynccnt)
  __builtin_amdgcn_s_wait_asynccnt(0);
#else
  asm volatile("s_wait_asynccnt 0x0" ::: "memory");
#endif
}

// ---------------- init: out = bias, absmax = 0 ----------------
__global__ void init_kernel(const float* __restrict__ bias, float* __restrict__ out,
                            unsigned* __restrict__ absmax) {
  int idx = blockIdx.x * blockDim.x + threadIdx.x;
  if (idx == 0) *absmax = 0u;
  out[idx] = bias[idx & (OUT_DIM - 1)];
}

// ---------------- global max|x| (non-negative -> raw bits monotonic) ----------------
__global__ __launch_bounds__(256) void absmax_kernel(const float* __restrict__ x,
                                                     unsigned* __restrict__ absmax) {
  float m = 0.f;
  for (int i = blockIdx.x * blockDim.x + threadIdx.x; i < T_DIM * IN_DIM;
       i += gridDim.x * blockDim.x)
    m = fmaxf(m, fabsf(x[i]));
  for (int off = 16; off; off >>= 1) m = fmaxf(m, __shfl_xor(m, off, 32));
  __shared__ float sm[8];
  int wave = threadIdx.x >> 5, lane = threadIdx.x & 31;
  if (lane == 0) sm[wave] = m;
  __syncthreads();
  if (threadIdx.x == 0) {
    for (int w = 1; w < 8; ++w) m = fmaxf(m, sm[w]);
    atomicMax(absmax, __float_as_uint(m));
  }
}

// ---------------- activation quantize + signed bit-plane extraction (fp16) ----------------
__global__ __launch_bounds__(256) void abits_kernel(const float* __restrict__ x,
                                                    const unsigned* __restrict__ absmax,
                                                    _Float16* __restrict__ xbits) {
  int idx = blockIdx.x * blockDim.x + threadIdx.x;
  float am = __uint_as_float(*absmax);
  float a_scale = (am > 0.f) ? am * (1.0f / QMAXI) : 1.0f;
  int q = (int)rintf(x[idx] / a_scale);
  #pragma unroll
  for (int z = 0; z < ZB; ++z) {
    int b = q % 2;               // signed remainder in {-1,0,1}
    q = (q - b) / 2;
    xbits[(size_t)z * (T_DIM * IN_DIM) + idx] = (_Float16)b;
  }
}

// ---------------- weight quantize per (row,subarray) + 7 cell slices ----------------
__global__ __launch_bounds__(256) void wq_kernel(const float* __restrict__ weight,
                                                 float* __restrict__ wscale,
                                                 _Float16* __restrict__ wrem) {
  int wave = threadIdx.x >> 5, lane = threadIdx.x & 31;
  int rid = blockIdx.x * 8 + wave;          // (o,n) pair, 8192 total
  int o = rid >> 3, n = rid & 7;
  const float4 w4 = *(const float4*)(&weight[(size_t)o * IN_DIM + n * SUBD + lane * 4]);
  float m = fmaxf(fmaxf(fabsf(w4.x), fabsf(w4.y)), fmaxf(fabsf(w4.z), fabsf(w4.w)));
  for (int off = 16; off; off >>= 1) m = fmaxf(m, __shfl_xor(m, off, 32));
  float scale = (m > 0.f) ? m * (1.0f / QMAXI) : 1.0f;
  if (lane == 0) wscale[o * NS + n] = scale;
  float wv[4] = {w4.x, w4.y, w4.z, w4.w};
  #pragma unroll
  for (int e = 0; e < 4; ++e) {
    int q = (int)rintf(wv[e] / scale);
    size_t col = (size_t)o * IN_DIM + n * SUBD + lane * 4 + e;
    #pragma unroll
    for (int k = 0; k < KS; ++k) {
      int r = q % 2;             // signed cell slice {-1,0,1}
      q = (q - r) / 2;
      wrem[(size_t)k * (OUT_DIM * IN_DIM) + col] = (_Float16)r;
    }
  }
}

// ---------------- per-(z,k): conductance noise + SAF -> fp16 B operand; reset minmax ----------------
__global__ __launch_bounds__(256) void rq_prep_kernel(const _Float16* __restrict__ wrem_k,
                                                      const unsigned char* __restrict__ sa0_k,
                                                      const unsigned char* __restrict__ sa1_k,
                                                      _Float16* __restrict__ rq,
                                                      unsigned* __restrict__ minmax, int zk) {
  if (blockIdx.x == 0 && threadIdx.x < 2) minmax[threadIdx.x] = threadIdx.x ? 0u : 0xFFFFFFFFu;
  int idx = blockIdx.x * blockDim.x + threadIdx.x;
  float rem = (float)wrem_k[idx];
  unsigned s = (unsigned)zk * 0x9E3779B9u;
  unsigned h1 = pcg((unsigned)idx * 2u + 1u + s);
  unsigned h2 = pcg((unsigned)idx * 2u + 2u + s);
  float u1 = (float)(h1 + 1u) * 2.3283064e-10f;   // (0,1]
  float u2 = (float)h2 * 2.3283064e-10f;
  float g = sqrtf(-2.f * __logf(u1)) * __cosf(6.2831853f * u2);
  float v = rem * (1.f + 0.05f * g);
  if (sa0_k[idx]) v = 0.1f;                                // stuck-at-0 -> 1/ONOFF
  else if (sa1_k[idx]) v = (v > 0.f) ? 1.f : ((v < 0.f) ? -1.f : 0.f);
  rq[idx] = (_Float16)v;
}

// ---------------- WMMA GEMM: part[n][t][o] = sum_s xbits_z[t][n*128+s] * rq[o][n*128+s] ----------------
__global__ __launch_bounds__(256)
void gemm_part_kernel(const _Float16* __restrict__ xbits_z,   // [T][IN]
                      const _Float16* __restrict__ rq,        // [OUT][IN]
                      float* __restrict__ part) {             // [NS][T][OUT]
  __shared__ __align__(16) _Float16 As[128 * LDA];
  __shared__ __align__(16) _Float16 Bs[128 * LDA];
  const int o0 = blockIdx.x * 128;
  const int t0 = blockIdx.y * 128;
  const int n  = blockIdx.z;
  const int tid = threadIdx.x;
  // stage 128x128 fp16 tiles via CDNA5 async global->LDS (no VGPR round-trip)
  for (int i = tid; i < 128 * 16; i += 256) {
    int r = i >> 4, c = (i & 15) * 8;
    async_g2l_b128((unsigned)(uintptr_t)&As[r * LDA + c],
                   &xbits_z[(size_t)(t0 + r) * IN_DIM + n * SUBD + c]);
    async_g2l_b128((unsigned)(uintptr_t)&Bs[r * LDA + c],
                   &rq[(size_t)(o0 + r) * IN_DIM + n * SUBD + c]);
  }
  wait_async0();
  __syncthreads();
  const int wave = tid >> 5, lane = tid & 31;
  const int wT = wave & 3, wO = wave >> 2;           // 4x2 wave grid over 128x128
  const int lh = lane >> 4, ll = lane & 15;
  v8f acc[2][4] = {};
  for (int kk = 0; kk < 128; kk += 32) {
    v16h a[2], b[4];
    #pragma unroll
    for (int mt = 0; mt < 2; ++mt) {
      int m = wT * 32 + mt * 16 + ll;
      v8h lo = *(const v8h*)(&As[m * LDA + kk + lh * 8]);        // K = kk + lh*8 .. +7
      v8h hi = *(const v8h*)(&As[m * LDA + kk + 16 + lh * 8]);   // K = kk+16+lh*8 .. +7
      a[mt] = __builtin_shufflevector(lo, hi, 0,1,2,3,4,5,6,7,8,9,10,11,12,13,14,15);
    }
    #pragma unroll
    for (int nt = 0; nt < 4; ++nt) {
      int o = wO * 64 + nt * 16 + ll;
      v8h lo = *(const v8h*)(&Bs[o * LDA + kk + lh * 8]);        // K = kk + lh*8 .. +7
      v8h hi = *(const v8h*)(&Bs[o * LDA + kk + 16 + lh * 8]);   // K = kk+16+lh*8 .. +7
      b[nt] = __builtin_shufflevector(lo, hi, 0,1,2,3,4,5,6,7,8,9,10,11,12,13,14,15);
    }
    #pragma unroll
    for (int mt = 0; mt < 2; ++mt)
      #pragma unroll
      for (int nt = 0; nt < 4; ++nt)
        acc[mt][nt] = __builtin_amdgcn_wmma_f32_16x16x32_f16(
            false, a[mt], false, b[nt], (short)0, acc[mt][nt], false, false);
  }
  float* pn = part + (size_t)n * (T_DIM * OUT_DIM);
  #pragma unroll
  for (int mt = 0; mt < 2; ++mt)
    #pragma unroll
    for (int nt = 0; nt < 4; ++nt) {
      int ob = o0 + wO * 64 + nt * 16 + ll;
      int mb = t0 + wT * 32 + mt * 16 + lh * 8;   // C layout: VGPR v -> row lh*8+v
      #pragma unroll
      for (int v = 0; v < 8; ++v)
        pn[(size_t)(mb + v) * OUT_DIM + ob] = acc[mt][nt][v];
    }
}

// ---------------- global min/max over part[NS][T][OUT] (ADC is global in reference) ----------------
__global__ __launch_bounds__(256) void minmax_kernel(const float* __restrict__ part,
                                                     unsigned* __restrict__ minmax) {
  const float4* p = (const float4*)part;
  const int N4 = NS * T_DIM * OUT_DIM / 4;
  float mn = 3.4e38f, mx = -3.4e38f;
  for (int i = blockIdx.x * blockDim.x + threadIdx.x; i < N4; i += gridDim.x * blockDim.x) {
    float4 v = p[i];
    mn = fminf(mn, fminf(fminf(v.x, v.y), fminf(v.z, v.w)));
    mx = fmaxf(mx, fmaxf(fmaxf(v.x, v.y), fmaxf(v.z, v.w)));
  }
  for (int off = 16; off; off >>= 1) {
    mn = fminf(mn, __shfl_xor(mn, off, 32));
    mx = fmaxf(mx, __shfl_xor(mx, off, 32));
  }
  __shared__ float smn[8], smx[8];
  int wave = threadIdx.x >> 5, lane = threadIdx.x & 31;
  if (lane == 0) { smn[wave] = mn; smx[wave] = mx; }
  __syncthreads();
  if (threadIdx.x == 0) {
    for (int w = 1; w < 8; ++w) { mn = fminf(mn, smn[w]); mx = fmaxf(mx, smx[w]); }
    atomicMin(&minmax[0], fenc(mn));
    atomicMax(&minmax[1], fenc(mx));
  }
}

// ---------------- ADC quantize + weight scale + accumulate into out ----------------
__global__ __launch_bounds__(256) void adc_accum_kernel(const float* __restrict__ part,
                                                        const unsigned* __restrict__ minmax,
                                                        const float* __restrict__ wscale,
                                                        const unsigned* __restrict__ absmax,
                                                        float* __restrict__ out, float zk_pow) {
  int idx = blockIdx.x * blockDim.x + threadIdx.x;   // t*OUT + o
  int o = idx & (OUT_DIM - 1);
  float mn = fdec(minmax[0]), mx = fdec(minmax[1]);
  float delta = mx - mn;
  float step = delta * 0.03125f;                     // 2^-ADC, ADC=5
  float rstep = (step > 0.f) ? 1.f / step : 0.f;
  float am = __uint_as_float(*absmax);
  float a_scale = (am > 0.f) ? am * (1.0f / QMAXI) : 1.0f;
  float sum = 0.f;
  #pragma unroll
  for (int n = 0; n < NS; ++n) {
    float v = part[(size_t)n * (T_DIM * OUT_DIM) + idx];
    float q = v;
    if (delta > 0.f) {
      float fi = fminf(fmaxf(floorf((v - mn) * rstep), 0.f), 31.f);
      q = fi * step + mn;
    }
    sum += q * wscale[o * NS + n];
  }
  out[idx] += sum * zk_pow * a_scale;
}

// ---------------- launch ----------------
extern "C" void kernel_launch(void* const* d_in, const int* in_sizes, int n_in,
                              void* d_out, int out_size, void* d_ws, size_t ws_size,
                              hipStream_t stream) {
  (void)in_sizes; (void)n_in; (void)out_size; (void)ws_size;
  const float* x            = (const float*)d_in[0];
  const float* weight       = (const float*)d_in[1];
  const float* bias         = (const float*)d_in[2];
  const unsigned char* sa0  = (const unsigned char*)d_in[3];  // bool masks [KS][OUT][NS][SUB]
  const unsigned char* sa1  = (const unsigned char*)d_in[4];
  float* out = (float*)d_out;

  // workspace layout (~64 MB needed)
  char* ws = (char*)d_ws;
  unsigned*  absmax = (unsigned*)(ws + 0);
  unsigned*  minmax = (unsigned*)(ws + 256);                       // [min_enc, max_enc]
  float*     wscale = (float*)(ws + 1024);                         // [OUT][NS]
  _Float16*  xbits  = (_Float16*)(ws + 1024 + 131072);             // [ZB][T][IN]
  _Float16*  wrem   = xbits + (size_t)ZB * T_DIM * IN_DIM;         // [KS][OUT][IN]
  _Float16*  rq     = wrem + (size_t)KS * OUT_DIM * IN_DIM;        // [OUT][IN]
  float*     part   = (float*)(rq + (size_t)OUT_DIM * IN_DIM);     // [NS][T][OUT]

  init_kernel<<<(T_DIM * OUT_DIM) / 256, 256, 0, stream>>>(bias, out, absmax);
  absmax_kernel<<<512, 256, 0, stream>>>(x, absmax);
  abits_kernel<<<(T_DIM * IN_DIM) / 256, 256, 0, stream>>>(x, absmax, xbits);
  wq_kernel<<<(OUT_DIM * NS) / 8, 256, 0, stream>>>(weight, wscale, wrem);

  for (int z = 0; z < ZB; ++z) {
    for (int k = 0; k < KS; ++k) {
      int zk = z * KS + k;
      const size_t kstride = (size_t)OUT_DIM * IN_DIM;
      rq_prep_kernel<<<(OUT_DIM * IN_DIM) / 256, 256, 0, stream>>>(
          wrem + (size_t)k * kstride, sa0 + (size_t)k * kstride, sa1 + (size_t)k * kstride,
          rq, minmax, zk);
      gemm_part_kernel<<<dim3(OUT_DIM / 128, T_DIM / 128, NS), 256, 0, stream>>>(
          xbits + (size_t)z * T_DIM * IN_DIM, rq, part);
      minmax_kernel<<<512, 256, 0, stream>>>(part, minmax);
      float zk_pow = (float)(1u << (z + k));   // 2^z * CELL_RANGE^k, CELL_RANGE=2
      adc_accum_kernel<<<(T_DIM * OUT_DIM) / 256, 256, 0, stream>>>(
          part, minmax, wscale, absmax, out, zk_pow);
    }
  }
}
